// Actor_69372311765507
// MI455X (gfx1250) — matrix operand order repaired and checked by reading, not compiled, not run on hardware
//
#include <hip/hip_runtime.h>

#define BATCH   256
#define STEPS   100
#define DT      0.001f
#define CH      20           // steps per chunk
#define NCH     (STEPS/CH)   // 5 chunks

typedef __bf16 bf16x16 __attribute__((ext_vector_type(16)));
typedef float  f32x8   __attribute__((ext_vector_type(8)));

union FragBF { bf16x16 v; unsigned int u[8]; };

__device__ __forceinline__ unsigned short f2bf(float x) {
    union { float f; unsigned int u; } a; a.f = x;
    unsigned int r = a.u + 0x7FFFu + ((a.u >> 16) & 1u);   // RNE
    return (unsigned short)(r >> 16);
}

// ---------------------------------------------------------------------------
// bf16 GEMM with f32 accumulate: C = relu(A(MxK) * B^T(NxK stored [N][K]) + bias)
// mode 0: write bf16 H[M][N].  mode 1: scatter f32 to out[b][step0+t][col].
// Block = 256 threads (8 waves), block tile 128(M) x 128(N), K-step 32.
// Wave tile 32(M) x 64(N): waves in 4(M) x 2(N) grid; 2 A-frags, 4 B-frags,
// 8 WMMAs per K-step per wave (48 LDS u32 reads per 8 WMMAs).
// ---------------------------------------------------------------------------
__global__ __launch_bounds__(256)
void gemm_bf16_wmma(const unsigned short* __restrict__ A,
                    const unsigned short* __restrict__ B,
                    const float* __restrict__ bias,
                    unsigned short* __restrict__ Hout,
                    float* __restrict__ Fout,
                    int M, int N, int K, int mode, int step0)
{
    __shared__ unsigned int ldsA[128 * 17];   // 128 rows x 32 bf16, stride 34 halves
    __shared__ unsigned int ldsB[128 * 17];

    const int tid   = threadIdx.x;
    const int lane  = tid & 31;
    const int wid   = tid >> 5;
    const int wid_m = wid >> 1;       // 0..3  -> 32-row stripe
    const int wid_n = wid & 1;        // 0..1  -> 64-col stripe
    const int m0    = blockIdx.y * 128;
    const int n0    = blockIdx.x * 128;

    f32x8 acc[8];                     // [mi][ni] -> acc[mi*4+ni]
#pragma unroll
    for (int i = 0; i < 8; i++)
#pragma unroll
        for (int j = 0; j < 8; j++) acc[i][j] = 0.0f;

    const uint4* A4 = (const uint4*)A;
    const uint4* B4 = (const uint4*)B;

    for (int k0 = 0; k0 < K; k0 += 32) {
        __syncthreads();
        // Stage 128x32 bf16 tiles of A and B: 512 uint4 each, 2 per thread.
#pragma unroll
        for (int j = 0; j < 2; j++) {
            int idx = j * 256 + tid;           // 0..511
            int r   = idx >> 2;                // tile row
            int q   = idx & 3;                 // 16B chunk within row
            uint4 va = A4[(((size_t)(m0 + r) * K + (size_t)k0) >> 3) + q];
            int lb = r * 17 + q * 4;
            ldsA[lb + 0] = va.x; ldsA[lb + 1] = va.y;
            ldsA[lb + 2] = va.z; ldsA[lb + 3] = va.w;
            uint4 vb = B4[(((size_t)(n0 + r) * K + (size_t)k0) >> 3) + q];
            ldsB[lb + 0] = vb.x; ldsB[lb + 1] = vb.y;
            ldsB[lb + 2] = vb.z; ldsB[lb + 3] = vb.w;
        }
        __syncthreads();

        // Prefetch next K-tile rows (global_prefetch_b8) to overlap compute.
        {
            int knext = k0 + 32;
            if (knext < K) {
                int r = tid & 127;
                const unsigned short* p = (tid < 128)
                    ? A + (size_t)(m0 + r) * K + knext
                    : B + (size_t)(n0 + r) * K + knext;
                __builtin_prefetch(p, 0, 1);
            }
        }

        // A fragments (16x32, ISA 7.12.2): lane L row = L&15;
        // uint idx per VGPR v: (v<4 ? v : v+4) + 4*(lane>=16)
        FragBF af[2];
        const int khA = (lane >> 4) << 2;
#pragma unroll
        for (int mi = 0; mi < 2; mi++) {
            const int rowA = wid_m * 32 + mi * 16 + (lane & 15);
#pragma unroll
            for (int v = 0; v < 8; v++)
                af[mi].u[v] = ldsA[rowA * 17 + ((v < 4) ? v : v + 4) + khA];
        }

        // B fragments (32x16): lane L column = L&15; same K order as A:
        // uint idx per VGPR v: (v<4 ? v : v+4) + 4*(lane>=16). B stored [N][K].
        const int khB = (lane >> 4) << 2;
#pragma unroll
        for (int ni = 0; ni < 4; ni++) {
            FragBF bfr;
            const int rowB = wid_n * 64 + ni * 16 + (lane & 15);
#pragma unroll
            for (int v = 0; v < 8; v++)
                bfr.u[v] = ldsB[rowB * 17 + ((v < 4) ? v : v + 4) + khB];
#pragma unroll
            for (int mi = 0; mi < 2; mi++)
                acc[mi * 4 + ni] = __builtin_amdgcn_wmma_f32_16x16x32_bf16(
                    false, af[mi].v, false, bfr.v, (short)0,
                    acc[mi * 4 + ni], false, false);
        }
    }

    // Epilogue. C/D layout: VGPR r -> M=r (lanes 0-15) / M=8+r (lanes 16-31).
    const int laneN = lane & 15;
    const int rofs  = (lane >> 4) << 3;
#pragma unroll
    for (int mi = 0; mi < 2; mi++) {
#pragma unroll
        for (int ni = 0; ni < 4; ni++) {
            const int col = n0 + wid_n * 64 + ni * 16 + laneN;
            const float bv = bias[col];
#pragma unroll
            for (int r = 0; r < 8; r++) {
                const int grow = m0 + wid_m * 32 + mi * 16 + rofs + r;
                float v = fmaxf(acc[mi * 4 + ni][r] + bv, 0.0f);
                if (mode == 0) {
                    Hout[(size_t)grow * N + col] = f2bf(v);
                } else {
                    const int t = step0 + (grow >> 8);     // row = t_local*256 + b
                    const int b = grow & 255;
                    Fout[((size_t)b * STEPS + t) * 256 + col] = v;
                }
            }
        }
    }
}

// ---------------------------------------------------------------------------
// Hopf oscillator chain: advance nsteps, emit bf16 states [t][b][1024].
// ---------------------------------------------------------------------------
__global__ __launch_bounds__(256)
void hopf_chunk(const float* __restrict__ z_in, float* __restrict__ z_state,
                const float* __restrict__ omega, const float* __restrict__ mu,
                const float* __restrict__ bb, float* __restrict__ z_out,
                unsigned short* __restrict__ Zbf, int step0, int nsteps)
{
    int idx = blockIdx.x * 256 + threadIdx.x;   // 256*512 threads
    int b = idx >> 9;
    int i = idx & 511;
    float x, y;
    if (step0 == 0) {
        x = z_in[(size_t)b * 1024 + i];
        y = z_in[(size_t)b * 1024 + 512 + i];
    } else {
        x = z_state[(size_t)b * 1024 + i];
        y = z_state[(size_t)b * 1024 + 512 + i];
    }
    const float m  = mu[(size_t)b * 512 + i];
    const float bv = bb[(size_t)b * 512 + i];
    const float w  = omega[b] * (float)(i + 1);
    for (int t = 0; t < nsteps; t++) {
        float r2 = x * x + y * y;
        float nx = x + DT * ((m - r2) * x - w * y + bv);
        float ny = y + DT * ((m - r2) * y + w * x);
        x = nx; y = ny;
        if (step0 == 0 && t == 0) {            // z_out = state after first step
            z_out[(size_t)b * 1024 + i]       = x;
            z_out[(size_t)b * 1024 + 512 + i] = y;
        }
        size_t row = ((size_t)t * 256 + b) * 1024;
        Zbf[row + i]       = f2bf(x);
        Zbf[row + 512 + i] = f2bf(y);
    }
    z_state[(size_t)b * 1024 + i]       = x;
    z_state[(size_t)b * 1024 + 512 + i] = y;
}

// ---------------------------------------------------------------------------
// Preamble (small fp32 GEMMs) + weight preparation
// ---------------------------------------------------------------------------
__global__ void cat_kernel(const float* __restrict__ ms, const float* __restrict__ rs,
                           const float* __restrict__ Wms, const float* __restrict__ bms,
                           const float* __restrict__ Wrs, const float* __restrict__ brs,
                           float* __restrict__ cat)
{
    int idx = blockIdx.x * 256 + threadIdx.x;   // 256*512
    int b = idx >> 9, j = idx & 511;
    float s;
    if (j < 256) {
        s = bms[j];
        for (int k = 0; k < 64; k++) s += ms[b * 64 + k] * Wms[k * 256 + j];
    } else {
        int jj = j - 256;
        s = brs[jj];
        for (int k = 0; k < 128; k++) s += rs[b * 128 + k] * Wrs[k * 256 + jj];
    }
    cat[idx] = fmaxf(s, 0.0f);
}

__global__ void st_kernel(const float* __restrict__ cat, const float* __restrict__ Wc,
                          const float* __restrict__ bc, float* __restrict__ st)
{
    int idx = blockIdx.x * 256 + threadIdx.x;
    int b = idx >> 9, j = idx & 511;
    float s = bc[j];
    for (int k = 0; k < 512; k++) s += cat[b * 512 + k] * Wc[k * 512 + j];
    st[idx] = fmaxf(s, 0.0f);
}

__global__ void mub_kernel(const float* __restrict__ st,
                           const float* __restrict__ Wmu, const float* __restrict__ bmu,
                           const float* __restrict__ Wb,  const float* __restrict__ bbv,
                           float* __restrict__ mu_out, float* __restrict__ bb_out)
{
    int idx = blockIdx.x * 256 + threadIdx.x;
    int b = idx >> 9, j = idx & 511;
    float s1 = bmu[j], s2 = bbv[j];
    for (int k = 0; k < 512; k++) {
        float a = st[b * 512 + k];
        s1 += a * Wmu[k * 512 + j];
        s2 += a * Wb[k * 512 + j];
    }
    mu_out[idx] = fmaxf(s1, 0.0f);   // abs(relu(x)) == relu(x)
    bb_out[idx] = fmaxf(s2, 0.0f);
}

__global__ void om_kernel(const float* __restrict__ st, const float* __restrict__ Wom,
                          const float* __restrict__ bom, float* __restrict__ om)
{
    int b = threadIdx.x;
    float s = bom[0];
    for (int k = 0; k < 512; k++) s += st[b * 512 + k] * Wom[k];
    om[b] = fmaxf(s, 0.0f);
}

// Build combined, transposed bf16 weight Wt[N2][K2] from Wr/Wi (fan_in x fan_out):
// rows n<n_out -> yr: [Wr ; -Wi],  rows n>=n_out -> yi: [Wi ; Wr]
__global__ void combine_w(const float* __restrict__ Wr, const float* __restrict__ Wi,
                          unsigned short* __restrict__ Wt, int n_in, int n_out, int has_yi)
{
    int K2 = 2 * n_in;
    long long total = (long long)n_out * (1 + has_yi) * K2;
    long long idx = (long long)blockIdx.x * 256 + threadIdx.x;
    if (idx >= total) return;
    int n = (int)(idx / K2), k = (int)(idx % K2);
    float v;
    if (n < n_out)
        v = (k < n_in) ?  Wr[(size_t)k * n_out + n] : -Wi[(size_t)(k - n_in) * n_out + n];
    else {
        int nn = n - n_out;
        v = (k < n_in) ?  Wi[(size_t)k * n_out + nn] :  Wr[(size_t)(k - n_in) * n_out + nn];
    }
    Wt[idx] = f2bf(v);
}

__global__ void combine_b(const float* __restrict__ br, const float* __restrict__ bi,
                          float* __restrict__ out, int n_out, int has_yi)
{
    int total = n_out * (1 + has_yi);
    int idx = blockIdx.x * 256 + threadIdx.x;
    if (idx >= total) return;
    out[idx] = (idx < n_out) ? br[idx] : bi[idx - n_out];
}

// ---------------------------------------------------------------------------
extern "C" void kernel_launch(void* const* d_in, const int* in_sizes, int n_in,
                              void* d_out, int out_size, void* d_ws, size_t ws_size,
                              hipStream_t stream)
{
    const float* motion = (const float*)d_in[0];
    const float* robot  = (const float*)d_in[1];
    const float* z      = (const float*)d_in[2];
    const float* W_ms = (const float*)d_in[3];  const float* b_ms = (const float*)d_in[4];
    const float* W_rs = (const float*)d_in[5];  const float* b_rs = (const float*)d_in[6];
    const float* W_cmb= (const float*)d_in[7];  const float* b_cmb= (const float*)d_in[8];
    const float* W_om = (const float*)d_in[9];  const float* b_om = (const float*)d_in[10];
    const float* W_mu = (const float*)d_in[11]; const float* b_mu = (const float*)d_in[12];
    const float* W_b  = (const float*)d_in[13]; const float* b_b  = (const float*)d_in[14];
    const float* Wr0  = (const float*)d_in[15]; const float* Wi0  = (const float*)d_in[16];
    const float* br0  = (const float*)d_in[17]; const float* bi0  = (const float*)d_in[18];
    const float* Wr1  = (const float*)d_in[19]; const float* Wi1  = (const float*)d_in[20];
    const float* br1  = (const float*)d_in[21]; const float* bi1  = (const float*)d_in[22];
    const float* Wr2  = (const float*)d_in[23]; const float* Wi2  = (const float*)d_in[24];
    const float* br2  = (const float*)d_in[25]; const float* bi2  = (const float*)d_in[26];

    // d_out: out(256*100*256) | z_out(256*1024) | omega(256) | mu(256*512) | bb(256*512)
    float* out     = (float*)d_out;
    float* z_out_p = out + (size_t)BATCH * STEPS * 256;
    float* omega_p = z_out_p + (size_t)BATCH * 1024;
    float* mu_p    = omega_p + BATCH;
    float* bb_p    = mu_p + (size_t)BATCH * 512;

    // Workspace carve-out (~53 MB)
    char* wp = (char*)d_ws;
    auto carve = [&](size_t bytes) { void* p = wp; wp += (bytes + 255) & ~(size_t)255; return p; };
    float* cat   = (float*)carve((size_t)256 * 512 * 4);
    float* st    = (float*)carve((size_t)256 * 512 * 4);
    float* bias0 = (float*)carve(2048 * 4);
    float* bias1 = (float*)carve(1024 * 4);
    float* bias2 = (float*)carve(256 * 4);
    unsigned short* Wc0t = (unsigned short*)carve((size_t)2048 * 1024 * 2);
    unsigned short* Wc1t = (unsigned short*)carve((size_t)1024 * 2048 * 2);
    unsigned short* Wc2t = (unsigned short*)carve((size_t)256  * 1024 * 2);
    float* z_state       = (float*)carve((size_t)256 * 1024 * 4);
    unsigned short* Zbf  = (unsigned short*)carve((size_t)CH * 256 * 1024 * 2);
    unsigned short* H0   = (unsigned short*)carve((size_t)CH * 256 * 2048 * 2);
    unsigned short* H1   = (unsigned short*)carve((size_t)CH * 256 * 1024 * 2);

    // Preamble
    cat_kernel<<<512, 256, 0, stream>>>(motion, robot, W_ms, b_ms, W_rs, b_rs, cat);
    st_kernel <<<512, 256, 0, stream>>>(cat, W_cmb, b_cmb, st);
    mub_kernel<<<512, 256, 0, stream>>>(st, W_mu, b_mu, W_b, b_b, mu_p, bb_p);
    om_kernel <<<1,   256, 0, stream>>>(st, W_om, b_om, omega_p);

    // Combined bf16 transposed weights + biases
    combine_w<<<(2048 * 1024 + 255) / 256, 256, 0, stream>>>(Wr0, Wi0, Wc0t, 512, 1024, 1);
    combine_w<<<(1024 * 2048 + 255) / 256, 256, 0, stream>>>(Wr1, Wi1, Wc1t, 1024, 512, 1);
    combine_w<<<(256  * 1024 + 255) / 256, 256, 0, stream>>>(Wr2, Wi2, Wc2t, 512, 256, 0);
    combine_b<<<8, 256, 0, stream>>>(br0, bi0, bias0, 1024, 1);
    combine_b<<<4, 256, 0, stream>>>(br1, bi1, bias1, 512, 1);
    combine_b<<<1, 256, 0, stream>>>(br2, bi2, bias2, 256, 0);

    // Chunked: hopf chain -> 3 WMMA GEMMs (M = CH*256 rows per chunk)
    const int Mc = CH * 256;                       // 5120
    dim3 g0(2048 / 128, Mc / 128);
    dim3 g1(1024 / 128, Mc / 128);
    dim3 g2(256  / 128, Mc / 128);
    for (int c = 0; c < NCH; c++) {
        int step0 = c * CH;
        hopf_chunk<<<512, 256, 0, stream>>>(z, z_state, omega_p, mu_p, bb_p,
                                            z_out_p, Zbf, step0, CH);
        gemm_bf16_wmma<<<g0, 256, 0, stream>>>(Zbf, Wc0t, bias0, H0, nullptr,
                                               Mc, 2048, 1024, 0, 0);
        gemm_bf16_wmma<<<g1, 256, 0, stream>>>(H0, Wc1t, bias1, H1, nullptr,
                                               Mc, 1024, 2048, 0, 0);
        gemm_bf16_wmma<<<g2, 256, 0, stream>>>(H1, Wc2t, bias2, nullptr, out,
                                               Mc, 256, 1024, 1, step0);
    }
}
